// FusedMoEBlockedF8_82712480186872
// MI455X (gfx1250) — matrix operand order repaired and checked by reading, not compiled, not run on hardware
//
#include <hip/hip_runtime.h>
#include <hip/hip_bf16.h>
#include <cstdint>
#include <cstddef>

// Problem constants (reference: T=2048, H=2048, F=1024, E=8, K=2, BLOCK=128)
#define TOK   2048
#define HID   2048
#define FDIM  1024
#define NEXP  8
#define KTOP  2
#define NSLOT (TOK * KTOP)
#define MBLK  128                          // rows per GEMM block

typedef __bf16 v16bf __attribute__((ext_vector_type(16)));
typedef float  v8f   __attribute__((ext_vector_type(8)));
typedef unsigned int u32x4 __attribute__((ext_vector_type(4)));
typedef float  f32x4 __attribute__((ext_vector_type(4)));
typedef __bf16 bf16x4 __attribute__((ext_vector_type(4)));

union AFrag { v16bf v; u32x4 q[2]; };     // 16x32 bf16 A fragment (per lane: 16 vals)
union BFrag { v16bf v; bf16x4 s[4]; };    // 32x16 bf16 B fragment (per lane: 16 vals)
union SPack { u32x4 q; bf16x4 h2[2]; };   // 8 bf16 staged as one 16B LDS store

// ---- workspace layout (bytes) ----
#define WS_OFF   0                         // int[NEXP]  expert row offsets
#define WS_CNT   64                        // int[NEXP]  expert row counts
#define WS_PTOK  128                       // int[NSLOT] slot -> token
#define WS_PW    (128 + NSLOT * 4)         // float[NSLOT] slot -> topk weight
#define WS_INV   (128 + NSLOT * 8)         // int[NSLOT] (t*K+k) -> slot
#define WS_ACT   (65536)                   // float[NSLOT*FDIM]   16 MB
#define WS_OUTE  (WS_ACT + (size_t)NSLOT * FDIM * 4)  // float[NSLOT*HID] 32 MB

// --------------------------------------------------------------------------
// Routing: histogram + exclusive prefix (single thread; E=8, NSLOT=4096)
// --------------------------------------------------------------------------
__global__ void k_hist(const long long* __restrict__ ids,
                       int* __restrict__ off, int* __restrict__ cnt) {
  if (threadIdx.x != 0 || blockIdx.x != 0) return;
  int c[NEXP];
  for (int e = 0; e < NEXP; ++e) c[e] = 0;
  for (int j = 0; j < NSLOT; ++j) c[(int)ids[j]]++;
  int acc = 0;
  for (int e = 0; e < NEXP; ++e) { off[e] = acc; cnt[e] = c[e]; acc += c[e]; }
}

// Deterministic stable sort by expert: slot = #{j : (e_j < e) || (e_j==e && j<idx)}
__global__ __launch_bounds__(256) void k_route(const long long* __restrict__ ids,
                                               const float* __restrict__ tw,
                                               int* __restrict__ ptok,
                                               float* __restrict__ pw,
                                               int* __restrict__ inv) {
  __shared__ int sid[1024];
  const int idx = blockIdx.x * 256 + threadIdx.x;
  const int e = (int)ids[idx];
  int slot = 0;
  for (int c0 = 0; c0 < NSLOT; c0 += 1024) {
    for (int j = threadIdx.x; j < 1024; j += 256) sid[j] = (int)ids[c0 + j];
    __syncthreads();
    for (int j = 0; j < 1024; ++j) {
      int ej = sid[j];
      int gj = c0 + j;
      slot += (ej < e) || (ej == e && gj < idx);
    }
    __syncthreads();
  }
  ptok[slot] = idx / KTOP;
  pw[slot]   = tw[idx];
  inv[idx]   = slot;
}

// --------------------------------------------------------------------------
// GEMM1: act[slot, f] = silu(x·Wg^T) * (x·Wu^T), block-scale fused into B.
// Block: 128 rows x 128 F-cols (gate + matching up panel). 8 waves; wave w
// owns 16 gate cols + the same 16 up cols. Per 32-K chunk each wave loads one
// gate-B + one up-B fragment and applies them to 8 shared A fragments
// -> 16 WMMAs per chunk per wave, B reuse x8.
// --------------------------------------------------------------------------
__global__ __launch_bounds__(256) void gemm1(
    const float* __restrict__ x,
    const float* __restrict__ wgu,     // (E, 2F, H)
    const float* __restrict__ sgu,     // (E, 2F/128=16, H/128=16)
    const int* __restrict__ off, const int* __restrict__ cnt,
    const int* __restrict__ ptok,
    float* __restrict__ act)           // (NSLOT, F)
{
  const int mt = blockIdx.x, nt = blockIdx.y, e = blockIdx.z;
  const int mrem = cnt[e] - mt * MBLK;
  if (mrem <= 0) return;
  const int base = off[e] + mt * MBLK;

  __shared__ __align__(16) unsigned short As[2][MBLK][40];  // double-buffered 128x32 bf16
  __shared__ int toks[MBLK];

  const int tid = threadIdx.x;
  if (tid < MBLK) toks[tid] = (tid < mrem) ? ptok[base + tid] : -1;
  __syncthreads();

  const int lane = tid & 31;
  const int wave = tid >> 5;
  const int half = lane >> 4;
  const int nl   = lane & 15;

  const size_t ocol = (size_t)nt * 128 + wave * 16 + nl;       // gate output col
  const float* wg_row = wgu + ((size_t)e * 2 * FDIM + ocol) * HID + 16 * half;
  const float* wu_row = wg_row + (size_t)FDIM * HID;           // up panel, +F rows
  const float* sgu_e  = sgu + (size_t)e * 16 * 16;

  // staging map: thread -> (row, 16-col span); 128x32 elems / 256 thr = 16 each
  const int sr = tid >> 1;
  const int sc = (tid & 1) * 16;
  const int stok = toks[sr];
  const float* xrow = (stok >= 0) ? (x + (size_t)stok * HID + sc) : nullptr;

  v8f cg[8] = {}, cu[8] = {};

  int p = 0;
  for (int h0 = 0; h0 < HID; h0 += 32, p ^= 1) {
    // stage A tile (f32 -> bf16, hardware pk-cvt) into LDS
    {
      SPack sp0, sp1;
      if (xrow) {
        const f32x4* px = (const f32x4*)(xrow + h0);
        sp0.h2[0] = __builtin_convertvector(px[0], bf16x4);
        sp0.h2[1] = __builtin_convertvector(px[1], bf16x4);
        sp1.h2[0] = __builtin_convertvector(px[2], bf16x4);
        sp1.h2[1] = __builtin_convertvector(px[3], bf16x4);
      } else {
        u32x4 z = {0u, 0u, 0u, 0u};
        sp0.q = z; sp1.q = z;
      }
      *(u32x4*)&As[p][sr][sc]     = sp0.q;
      *(u32x4*)&As[p][sr][sc + 8] = sp1.q;
    }
    __syncthreads();   // single barrier per chunk (double-buffered)

    const float s_g = sgu_e[nt * 16 + (h0 >> 7)];        // gate o-block = nt
    const float s_u = sgu_e[(8 + nt) * 16 + (h0 >> 7)];  // up   o-block = 8+nt

    // B fragments from global, scale folded in via packed mul + pk convert
    BFrag bg, bu;
    const f32x4* pg = (const f32x4*)(wg_row + h0);
    const f32x4* pu = (const f32x4*)(wu_row + h0);
#pragma unroll
    for (int j = 0; j < 4; ++j) {
      bg.s[j] = __builtin_convertvector(pg[j] * s_g, bf16x4);
      bu.s[j] = __builtin_convertvector(pu[j] * s_u, bf16x4);
    }
    if (h0 + 512 < HID) {
      __builtin_prefetch((const void*)(wg_row + h0 + 512), 0, 0);
      __builtin_prefetch((const void*)(wu_row + h0 + 512), 0, 0);
    }

#pragma unroll
    for (int f = 0; f < 8; ++f) {
      AFrag a;
      a.q[0] = *(const u32x4*)&As[p][16 * f + nl][16 * half];      // K 16h..16h+7
      a.q[1] = *(const u32x4*)&As[p][16 * f + nl][16 * half + 8];  // K 16h+8..16h+15
      cg[f] = __builtin_amdgcn_wmma_f32_16x16x32_bf16(false, a.v, false, bg.v,
                                                      (short)0, cg[f], false, false);
      cu[f] = __builtin_amdgcn_wmma_f32_16x16x32_bf16(false, a.v, false, bu.v,
                                                      (short)0, cu[f], false, false);
    }
  }

  // epilogue: silu(gate) * up -> act   (fast rcp instead of precise divide)
  const int fcol = nt * 128 + wave * 16 + nl;
#pragma unroll
  for (int f = 0; f < 8; ++f) {
#pragma unroll
    for (int i = 0; i < 8; ++i) {
      int r = 16 * f + i + 8 * half;          // C layout: M = i + 8*half
      if (r < mrem) {
        float g = cg[f][i], u = cu[f][i];
        float av = g * __builtin_amdgcn_rcpf(1.f + __expf(-g)) * u;
        act[(size_t)(base + r) * FDIM + fcol] = av;
      }
    }
  }
}

// --------------------------------------------------------------------------
// GEMM2: oute[slot, h] = topkw[slot] * (act[slot,:] · Wdn[e,h,:]), scale fused
// Block: 128 rows x 128 H-cols; 8 WMMAs per 32-K chunk per wave.
// --------------------------------------------------------------------------
__global__ __launch_bounds__(256) void gemm2(
    const float* __restrict__ act,     // (NSLOT, F)
    const float* __restrict__ wdn,     // (E, H, F)
    const float* __restrict__ sdn,     // (E, H/128=16, F/128=8)
    const int* __restrict__ off, const int* __restrict__ cnt,
    const float* __restrict__ pw,
    float* __restrict__ oute)          // (NSLOT, H)
{
  const int mt = blockIdx.x, nt = blockIdx.y, e = blockIdx.z;
  const int mrem = cnt[e] - mt * MBLK;
  if (mrem <= 0) return;
  const int base = off[e] + mt * MBLK;

  __shared__ __align__(16) unsigned short As[2][MBLK][40];
  __shared__ float wls[MBLK];

  const int tid = threadIdx.x;
  if (tid < MBLK) wls[tid] = (tid < mrem) ? pw[base + tid] : 0.f;
  __syncthreads();

  const int lane = tid & 31;
  const int wave = tid >> 5;
  const int half = lane >> 4;
  const int nl   = lane & 15;

  const size_t hcol = (size_t)nt * 128 + wave * 16 + nl;
  const float* wd_row = wdn + ((size_t)e * HID + hcol) * FDIM + 16 * half;
  const float* sdn_e  = sdn + (size_t)e * 16 * 8;

  const int sr = tid >> 1;
  const int sc = (tid & 1) * 16;
  const bool svalid = (sr < mrem);
  const float* arow = act + (size_t)(base + sr) * FDIM + sc;

  v8f c[8] = {};

  int p = 0;
  for (int f0 = 0; f0 < FDIM; f0 += 32, p ^= 1) {
    {
      SPack sp0, sp1;
      if (svalid) {
        const f32x4* pa = (const f32x4*)(arow + f0);
        sp0.h2[0] = __builtin_convertvector(pa[0], bf16x4);
        sp0.h2[1] = __builtin_convertvector(pa[1], bf16x4);
        sp1.h2[0] = __builtin_convertvector(pa[2], bf16x4);
        sp1.h2[1] = __builtin_convertvector(pa[3], bf16x4);
      } else {
        u32x4 z = {0u, 0u, 0u, 0u};
        sp0.q = z; sp1.q = z;
      }
      *(u32x4*)&As[p][sr][sc]     = sp0.q;
      *(u32x4*)&As[p][sr][sc + 8] = sp1.q;
    }
    __syncthreads();

    const float s = sdn_e[nt * 8 + (f0 >> 7)];

    BFrag b;
    const f32x4* pb = (const f32x4*)(wd_row + f0);
#pragma unroll
    for (int j = 0; j < 4; ++j)
      b.s[j] = __builtin_convertvector(pb[j] * s, bf16x4);
    if (f0 + 512 < FDIM)
      __builtin_prefetch((const void*)(wd_row + f0 + 512), 0, 0);

#pragma unroll
    for (int f = 0; f < 8; ++f) {
      AFrag a;
      a.q[0] = *(const u32x4*)&As[p][16 * f + nl][16 * half];
      a.q[1] = *(const u32x4*)&As[p][16 * f + nl][16 * half + 8];
      c[f] = __builtin_amdgcn_wmma_f32_16x16x32_bf16(false, a.v, false, b.v,
                                                     (short)0, c[f], false, false);
    }
  }

#pragma unroll
  for (int f = 0; f < 8; ++f) {
#pragma unroll
    for (int i = 0; i < 8; ++i) {
      int r = 16 * f + i + 8 * half;
      if (r < mrem) oute[(size_t)(base + r) * HID + hcol] = c[f][i] * wls[r];
    }
  }
}

// --------------------------------------------------------------------------
// Combine the K=2 expert contributions per token (weights already folded).
// --------------------------------------------------------------------------
__global__ __launch_bounds__(256) void k_combine(const float* __restrict__ oute,
                                                 const int* __restrict__ inv,
                                                 float* __restrict__ out) {
  const int idx = blockIdx.x * 256 + threadIdx.x;
  if (idx >= TOK * HID) return;
  const int t = idx / HID, h = idx % HID;
  const int s0 = inv[2 * t], s1 = inv[2 * t + 1];
  out[idx] = oute[(size_t)s0 * HID + h] + oute[(size_t)s1 * HID + h];
}

// --------------------------------------------------------------------------
extern "C" void kernel_launch(void* const* d_in, const int* in_sizes, int n_in,
                              void* d_out, int out_size, void* d_ws, size_t ws_size,
                              hipStream_t stream) {
  (void)in_sizes; (void)n_in; (void)out_size; (void)ws_size;
  const float*     x   = (const float*)d_in[0];      // hidden_states (T,H) f32
  const float*     tw  = (const float*)d_in[1];      // topk_weights  (T,K) f32
  const long long* ids = (const long long*)d_in[2];  // topk_ids      (T,K) i64
  const float*     wgu = (const float*)d_in[3];      // (E,2F,H) f32
  const float*     sgu = (const float*)d_in[4];      // (E,16,16) f32
  const float*     wdn = (const float*)d_in[5];      // (E,H,F) f32
  const float*     sdn = (const float*)d_in[6];      // (E,16,8) f32
  float* out = (float*)d_out;

  char* ws = (char*)d_ws;
  int*   off  = (int*)(ws + WS_OFF);
  int*   cnt  = (int*)(ws + WS_CNT);
  int*   ptok = (int*)(ws + WS_PTOK);
  float* pw   = (float*)(ws + WS_PW);
  int*   inv  = (int*)(ws + WS_INV);
  float* act  = (float*)(ws + WS_ACT);
  float* oute = (float*)(ws + WS_OUTE);

  k_hist<<<1, 1, 0, stream>>>(ids, off, cnt);
  k_route<<<NSLOT / 256, 256, 0, stream>>>(ids, tw, ptok, pw, inv);
  gemm1<<<dim3(NSLOT / MBLK, FDIM / 128, NEXP), 256, 0, stream>>>(x, wgu, sgu, off, cnt, ptok, act);
  gemm2<<<dim3(NSLOT / MBLK, HID / 128, NEXP), 256, 0, stream>>>(act, wdn, sdn, off, cnt, pw, oute);
  k_combine<<<(TOK * HID + 255) / 256, 256, 0, stream>>>(oute, inv, out);
}
